// GridMambaBlock_67731634257996
// MI455X (gfx1250) — matrix operand order repaired and checked by reading, not compiled, not run on hardware
//
#include <hip/hip_runtime.h>

typedef __attribute__((ext_vector_type(16))) _Float16 v16h;
typedef __attribute__((ext_vector_type(8)))  _Float16 v8h;
typedef __attribute__((ext_vector_type(8)))  float    v8f;

#define NROW  65536   // 512 sequences * 128 positions
#define LSEQ  128
#define NSEQ  512
#define DIN   386     // d_in_proj
#define NH    2

__device__ __forceinline__ float sigmoidf_(float x){ return 1.f/(1.f+__expf(-x)); }

// ---------------- weight conversion to f16 (once per launch, deterministic) --------
__global__ void prep_weights_k(const float* __restrict__ Wp, const float* __restrict__ Wo,
                               const float* __restrict__ fW,
                               _Float16* __restrict__ Wp16, _Float16* __restrict__ Wo16,
                               _Float16* __restrict__ fW16) {
  int i = blockIdx.x*blockDim.x + threadIdx.x;
  if (i < 4*386*64) Wp16[i] = (_Float16)Wp[i];
  if (i < 4*64*128) Wo16[i] = (_Float16)Wo[i];
  if (i < 2*64*128) fW16[i] = (_Float16)fW[i];
}

// ---------------- layernorm + gather transpose -> u16 (65536 x 64 f16) ------------
__global__ __launch_bounds__(64) void ln_k(const float* __restrict__ X,
                                           _Float16* __restrict__ U16,
                                           const float* __restrict__ w,
                                           const float* __restrict__ b, int stage) {
  int m = blockIdx.x, c = threadIdx.x;
  int bb = m >> 14, rem = m & 16383, s1 = rem >> 7, l = rem & 127;
  size_t xi = (stage==0) ? ((((size_t)bb*64 + c)*128 + s1)*128 + l)
                         : ((((size_t)bb*64 + c)*128 + l)*128 + s1);
  float v = X[xi];
  __shared__ float r1[64], r2[64];
  r1[c] = v; r2[c] = v*v; __syncthreads();
  for (int o = 32; o > 0; o >>= 1) {
    if (c < o) { r1[c] += r1[c+o]; r2[c] += r2[c+o]; }
    __syncthreads();
  }
  float mean = r1[0]*(1.f/64.f);
  float var  = r2[0]*(1.f/64.f) - mean*mean;
  float y = (v - mean)*rsqrtf(var + 1e-5f)*w[c] + b[c];
  U16[(size_t)m*64 + c] = (_Float16)y;
}

// ---------------- in_proj: (65536x64) @ (386x64)^T via WMMA f16 -------------------
// one wave per block; one A fragment reused across 4 n-tiles (4 accumulators).
// store region is tile-uniform: tiles 0-7 -> z16, 8-23 -> xbc, 24 -> dt raw
__global__ __launch_bounds__(32) void inproj_wmma_k(const _Float16* __restrict__ U,
                                                    const _Float16* __restrict__ W,
                                                    _Float16* __restrict__ Z16,
                                                    float* __restrict__ XBC,
                                                    float* __restrict__ DTRAW) {
  int lane = threadIdx.x;
  int m0 = blockIdx.x*16;
  int ng = blockIdx.y;                       // n-tile group: tiles 4*ng .. 4*ng+3
  int r = lane & 15, hi = lane >> 4;
  const _Float16* Ap = U + (size_t)(m0 + r)*64 + hi*16;
  v16h a0, a1;                               // K = 64 -> two 16x32 A fragments
  ((v8h*)&a0)[0] = *(const v8h*)(Ap +  0);
  ((v8h*)&a0)[1] = *(const v8h*)(Ap +  8);
  ((v8h*)&a1)[0] = *(const v8h*)(Ap + 32);
  ((v8h*)&a1)[1] = *(const v8h*)(Ap + 40);
  #pragma unroll
  for (int j = 0; j < 4; ++j) {
    int nt = ng*4 + j;
    if (nt >= 25) continue;                  // block-uniform guard (EXEC stays full)
    int nb = nt*16 + r;
    v16h b0 = {}, b1 = {};
    if (nb < DIN) {                          // per-lane guard on loads only
      const _Float16* Bp = W + (size_t)nb*64 + hi*16;
      b0 = *(const v16h*)(Bp);
      b1 = *(const v16h*)(Bp + 32);
    }
    v8f acc = {};
    acc = __builtin_amdgcn_wmma_f32_16x16x32_f16(false, a0, false, b0, (short)0, acc, false, false);
    acc = __builtin_amdgcn_wmma_f32_16x16x32_f16(false, a1, false, b1, (short)0, acc, false, false);
    int n = nt*16 + r;
    if (nt < 8) {                            // z region (tile-uniform)
      #pragma unroll
      for (int v = 0; v < 8; ++v)
        Z16[(size_t)(m0 + hi*8 + v)*128 + n] = (_Float16)acc[v];
    } else if (nt < 24) {                    // xBC region
      #pragma unroll
      for (int v = 0; v < 8; ++v)
        XBC[(size_t)(m0 + hi*8 + v)*256 + (n-128)] = acc[v];
    } else {                                 // dt region, only n = 384,385 valid
      if (n < DIN) {
        #pragma unroll
        for (int v = 0; v < 8; ++v)
          DTRAW[(size_t)(m0 + hi*8 + v)*2 + (n-384)] = acc[v];
      }
    }
  }
}

// ---------------- dt = softplus(raw + bias); dA = exp(-exp(A_log)*dt) --------------
__global__ void dt_k(const float* __restrict__ DTRAW, const float* __restrict__ dtb,
                     const float* __restrict__ Al, float* __restrict__ DT,
                     float* __restrict__ DA) {
  int i = blockIdx.x*blockDim.x + threadIdx.x;
  if (i >= NROW*NH) return;
  int h = i & 1;
  float raw = DTRAW[i] + dtb[h];
  float dt = (raw > 20.f) ? raw : log1pf(expf(raw));
  DT[i] = dt;
  DA[i] = expf(-expf(Al[h]) * dt);
}

// ---------------- depthwise causal conv (k=4) + SiLU, in-place, direction-aware ----
__global__ __launch_bounds__(256) void conv_k(float* __restrict__ XBC,
                                              const float* __restrict__ cw,
                                              const float* __restrict__ cb, int rev) {
  int s = blockIdx.x, c = threadIdx.x;
  float w0 = cw[c*4+0], w1 = cw[c*4+1], w2 = cw[c*4+2], w3 = cw[c*4+3], bias = cb[c];
  float xm1 = 0.f, xm2 = 0.f, xm3 = 0.f;          // register shift window
  long step = rev ? -256 : 256;
  for (int i = 0; i < LSEQ; ++i) {
    int t = rev ? (LSEQ-1-i) : i;
    size_t idx = ((size_t)s*LSEQ + t)*256 + c;
    if (i + 1 < LSEQ) __builtin_prefetch(XBC + (long)idx + step, 0, 1);
    float xc = XBC[idx];
    float o = fmaf(w0, xm3, fmaf(w1, xm2, fmaf(w2, xm1, fmaf(w3, xc, bias))));
    o = o * sigmoidf_(o);                          // silu
    XBC[idx] = o;
    xm3 = xm2; xm2 = xm1; xm1 = xc;
  }
}

// ---------------- selective scan: state 64x64 per (seq,head) in VGPRs --------------
// per-step B||C broadcast (512B) is pulled straight into LDS with
// GLOBAL_LOAD_ASYNC_TO_LDS_B64 (ASYNCcnt), bypassing VGPRs.
__global__ __launch_bounds__(64) void scan_k(const float* __restrict__ XBC,
                                             const float* __restrict__ DT,
                                             const float* __restrict__ DA,
                                             float* __restrict__ Y, int rev) {
  int s = blockIdx.x, h = blockIdx.y, p = threadIdx.x;
  __shared__ __align__(16) float BC[128];          // [0,64) = B_t, [64,128) = C_t
  float hreg[64];
  #pragma unroll
  for (int n = 0; n < 64; ++n) hreg[n] = 0.f;
  // wave-relative LDS byte address = low 32 bits of generic shared pointer
  unsigned ldsoff = (unsigned)(unsigned long long)(const void*)&BC[2*p];
  for (int i = 0; i < LSEQ; ++i) {
    int t = rev ? (LSEQ-1-i) : i;
    size_t row = (size_t)s*LSEQ + t;
    __syncthreads();                               // prior BC consumers done
    unsigned long long ga = (unsigned long long)(XBC + row*256 + 128 + 2*p);
    asm volatile("global_load_async_to_lds_b64 %0, %1, off"
                 :: "v"(ldsoff), "v"(ga) : "memory");
    asm volatile("s_wait_asynccnt 0x0" ::: "memory");
    __syncthreads();                               // both waves' chunks visible
    float xv  = XBC[row*256 + h*64 + p];
    float dtv = DT[row*2 + h];
    float dav = DA[row*2 + h];
    float a = dtv * xv;
    float y = 0.f;
    #pragma unroll
    for (int n = 0; n < 64; ++n) {
      hreg[n] = fmaf(hreg[n], dav, a*BC[n]);
      y = fmaf(hreg[n], BC[64+n], y);
    }
    Y[row*128 + h*64 + p] = y;
  }
}

// ---------------- y += D*x; gate with silu(z); RMSNorm; -> g16 ---------------------
__global__ __launch_bounds__(128) void gate_k(const float* __restrict__ Y,
                                              const float* __restrict__ XBC,
                                              const _Float16* __restrict__ Z16,
                                              const float* __restrict__ Dp,
                                              const float* __restrict__ nw,
                                              _Float16* __restrict__ G16) {
  int m = blockIdx.x, c = threadIdx.x, h = c >> 6;
  float y = Y[(size_t)m*128 + c] + Dp[h]*XBC[(size_t)m*256 + c];
  float z = (float)Z16[(size_t)m*128 + c];
  float g = y * (z * sigmoidf_(z));
  __shared__ float rs[128];
  rs[c] = g*g; __syncthreads();
  for (int o = 64; o > 0; o >>= 1) {
    if (c < o) rs[c] += rs[c+o];
    __syncthreads();
  }
  float r = rsqrtf(rs[0]*(1.f/128.f) + 1e-5f);
  G16[(size_t)m*128 + c] = (_Float16)(g*r*nw[c]);
}

// ---------------- K=128 WMMA GEMM, N=64: one wave, 1 A fragment x 4 n-tiles --------
// mode 0: out_proj -> cat16[:, colbase:colbase+64] (f16)
// mode 1: fusion   -> scatter-add residual into x layout (f32)
__global__ __launch_bounds__(32) void gemm128_wmma_k(const _Float16* __restrict__ U,
                                                     const _Float16* __restrict__ W,
                                                     _Float16* __restrict__ OUT16, int colbase,
                                                     const float* __restrict__ fb,
                                                     const float* __restrict__ XSRC,
                                                     float* __restrict__ XDST,
                                                     int stage, int mode) {
  int lane = threadIdx.x;
  int m0 = blockIdx.x*16;
  int r = lane & 15, hi = lane >> 4;
  const _Float16* Ap = U + (size_t)(m0 + r)*128 + hi*16;
  const _Float16* Bp = W + (size_t)r*128 + hi*16;
  v8f acc[4] = {};
  #pragma unroll
  for (int kc = 0; kc < 128; kc += 32) {
    v16h a;
    ((v8h*)&a)[0] = *(const v8h*)(Ap + kc);
    ((v8h*)&a)[1] = *(const v8h*)(Ap + kc + 8);
    #pragma unroll
    for (int j = 0; j < 4; ++j) {
      v16h b = *(const v16h*)(Bp + (size_t)j*16*128 + kc);
      acc[j] = __builtin_amdgcn_wmma_f32_16x16x32_f16(false, a, false, b, (short)0, acc[j], false, false);
    }
  }
  if (mode == 0) {
    #pragma unroll
    for (int j = 0; j < 4; ++j) {
      int n = j*16 + r;
      #pragma unroll
      for (int v = 0; v < 8; ++v)
        OUT16[(size_t)(m0 + hi*8 + v)*128 + colbase + n] = (_Float16)acc[j][v];
    }
  } else {
    #pragma unroll
    for (int j = 0; j < 4; ++j) {
      int n = j*16 + r;
      float bias = fb[n];
      #pragma unroll
      for (int v = 0; v < 8; ++v) {
        int m = m0 + hi*8 + v;
        int bb = m >> 14, rem = m & 16383, s1 = rem >> 7, l = rem & 127;
        size_t xi = (stage==0) ? ((((size_t)bb*64 + n)*128 + s1)*128 + l)
                               : ((((size_t)bb*64 + n)*128 + l)*128 + s1);
        XDST[xi] = XSRC[xi] + acc[j][v] + bias;    // residual add folded in
      }
    }
  }
}

extern "C" void kernel_launch(void* const* d_in, const int* in_sizes, int n_in,
                              void* d_out, int out_size, void* d_ws, size_t ws_size,
                              hipStream_t stream) {
  const float* x_in = (const float*)d_in[0];
  const float* Wp   = (const float*)d_in[1];
  const float* cw   = (const float*)d_in[2];
  const float* cb   = (const float*)d_in[3];
  const float* dtb  = (const float*)d_in[4];
  const float* Al   = (const float*)d_in[5];
  const float* Dp   = (const float*)d_in[6];
  const float* nw   = (const float*)d_in[7];
  const float* Wo   = (const float*)d_in[8];
  const float* fW   = (const float*)d_in[9];
  const float* fb   = (const float*)d_in[10];
  const float* lnw  = (const float*)d_in[11];
  const float* lnb  = (const float*)d_in[12];
  float* out = (float*)d_out;

  char* ws = (char*)d_ws;
  size_t off = 0;
  auto alloc = [&](size_t bytes) -> char* {
    char* p = ws + off; off += (bytes + 255) & ~(size_t)255; return p;
  };
  _Float16* u16   = (_Float16*)alloc((size_t)NROW*64*2);
  _Float16* Wp16  = (_Float16*)alloc((size_t)4*386*64*2);
  _Float16* Wo16  = (_Float16*)alloc((size_t)4*64*128*2);
  _Float16* fW16  = (_Float16*)alloc((size_t)2*64*128*2);
  _Float16* z16   = (_Float16*)alloc((size_t)NROW*128*2);
  float*    xbc   = (float*)   alloc((size_t)NROW*256*4);
  float*    dtraw = (float*)   alloc((size_t)NROW*2*4);
  float*    dtbuf = (float*)   alloc((size_t)NROW*2*4);
  float*    dabuf = (float*)   alloc((size_t)NROW*2*4);
  float*    ybuf  = (float*)   alloc((size_t)NROW*128*4);
  _Float16* g16   = (_Float16*)alloc((size_t)NROW*128*2);
  _Float16* cat16 = (_Float16*)alloc((size_t)NROW*128*2);
  float*    xwork = (float*)   alloc((size_t)4*64*128*128*4);

  prep_weights_k<<<(98816 + 255)/256, 256, 0, stream>>>(Wp, Wo, fW, Wp16, Wo16, fW16);

  for (int stage = 0; stage < 2; ++stage) {
    const float* xs = (stage == 0) ? x_in  : xwork;
    float*       xd = (stage == 0) ? xwork : out;
    ln_k<<<NROW, 64, 0, stream>>>(xs, u16, lnw + stage*64, lnb + stage*64, stage);
    for (int d = 0; d < 2; ++d) {                  // d=0 fwd, d=1 bwd (index-reversed)
      int pi = stage*2 + d;
      inproj_wmma_k<<<dim3(4096, 7), 32, 0, stream>>>(
          u16, Wp16 + (size_t)pi*386*64, z16, xbc, dtraw);
      dt_k<<<(NROW*2 + 255)/256, 256, 0, stream>>>(
          dtraw, dtb + pi*2, Al + pi*2, dtbuf, dabuf);
      conv_k<<<NSEQ, 256, 0, stream>>>(xbc, cw + (size_t)pi*256*4, cb + (size_t)pi*256, d);
      scan_k<<<dim3(NSEQ, 2), 64, 0, stream>>>(xbc, dtbuf, dabuf, ybuf, d);
      gate_k<<<NROW, 128, 0, stream>>>(ybuf, xbc, z16, Dp + pi*2, nw + (size_t)pi*128, g16);
      gemm128_wmma_k<<<4096, 32, 0, stream>>>(
          g16, Wo16 + (size_t)pi*64*128, cat16, d*64,
          (const float*)0, (const float*)0, (float*)0, 0, 0);
    }
    gemm128_wmma_k<<<4096, 32, 0, stream>>>(
        cat16, fW16 + (size_t)stage*64*128, (_Float16*)0, 0,
        fb + stage*64, xs, xd, stage, 1);
  }
}
